// PM25TransformerModel_58583353917570
// MI455X (gfx1250) — matrix operand order repaired and checked by reading, not compiled, not run on hardware
//
#include <hip/hip_runtime.h>
#include <math.h>

// ---------------- constants (match reference) ----------------
constexpr int Bc  = 32;
constexpr int Tc  = 512;
constexpr int Dc  = 384;
constexpr int Hc  = 8;
constexpr int HSc = 48;     // head size
constexpr int HSP = 64;     // padded head size (clean K=32 steps)
constexpr int FFc = 1536;
constexpr int Lc  = 6;
constexpr int Mc  = Bc * Tc;      // 16384 tokens
constexpr int NQK = Hc * HSP;     // 512 padded q/k/v width

typedef __attribute__((ext_vector_type(16))) __bf16 bf16x16;
typedef __attribute__((ext_vector_type(8)))  __bf16 bf16x8;
typedef __attribute__((ext_vector_type(8)))  float  f32x8;

#define DEV __device__ __forceinline__

DEV f32x8 zero8() {
    f32x8 z;
#pragma unroll
    for (int i = 0; i < 8; ++i) z[i] = 0.f;
    return z;
}

// A fragment (16x32, 16-bit): lane<16 -> K{0..7,16..23}; lane>=16 -> K{8..15,24..31}
DEV bf16x16 load_afrag(const __bf16* p, int half) {
    const __bf16* q = p + half * 8;
    bf16x8 lo = *reinterpret_cast<const bf16x8*>(q);
    bf16x8 hi = *reinterpret_cast<const bf16x8*>(q + 16);
    return __builtin_shufflevector(lo, hi, 0, 1, 2, 3, 4, 5, 6, 7,
                                   8, 9, 10, 11, 12, 13, 14, 15);
}

// B fragment (32x16, 16-bit): lane<16 -> K{0..7,16..23}; lane>=16 -> K{8..15,24..31}; p = BT row (contiguous K)
DEV bf16x16 load_bfrag(const __bf16* p, int half) {
    const __bf16* q = p + half * 8;
    bf16x8 lo = *reinterpret_cast<const bf16x8*>(q);
    bf16x8 hi = *reinterpret_cast<const bf16x8*>(q + 16);
    return __builtin_shufflevector(lo, hi, 0, 1, 2, 3, 4, 5, 6, 7,
                                   8, 9, 10, 11, 12, 13, 14, 15);
}

DEV f32x8 wmma_bf16(bf16x16 a, bf16x16 b, f32x8 c) {
    return __builtin_amdgcn_wmma_f32_16x16x32_bf16(false, a, false, b,
                                                   (short)0, c, false, false);
}

DEV float half16_max(float v) {
#pragma unroll
    for (int o = 1; o < 16; o <<= 1) v = fmaxf(v, __shfl_xor(v, o));
    return v;
}
DEV float half16_sum(float v) {
#pragma unroll
    for (int o = 1; o < 16; o <<= 1) v += __shfl_xor(v, o);
    return v;
}
DEV float wave32_sum(float v) {
#pragma unroll
    for (int o = 1; o < 32; o <<= 1) v += __shfl_xor(v, o);
    return v;
}

// ---------------- weight conversion ----------------
// generic: W[R][C] fp32 (row-major) -> WT[C][R] bf16
__global__ __launch_bounds__(256) void convT_kernel(const float* __restrict__ W,
                                                    __bf16* __restrict__ WT,
                                                    int R, int C) {
    int idx = blockIdx.x * 256 + threadIdx.x;
    if (idx >= R * C) return;
    int r = idx / C, c = idx - r * C;
    WT[(size_t)c * R + r] = (__bf16)W[idx];
}

// Wq/Wk/Wv for one layer: in (H,D,HS) -> out [H*HSP][D] transposed, padded with zeros
__global__ __launch_bounds__(256) void convQKV_kernel(const float* __restrict__ W,
                                                      __bf16* __restrict__ WT) {
    int idx = blockIdx.x * 256 + threadIdx.x;
    if (idx >= NQK * Dc) return;
    int n = idx / Dc, d = idx - n * Dc;
    int h = n >> 6, k = n & 63;
    float v = (k < HSc) ? W[((size_t)h * Dc + d) * HSc + k] : 0.f;
    WT[idx] = (__bf16)v;
}

// ---------------- embedding ----------------
__global__ __launch_bounds__(256) void embed_kernel(const float* __restrict__ pm25,
                                                    const int* __restrict__ hour,
                                                    const int* __restrict__ day,
                                                    const float* __restrict__ Wp,
                                                    const float* __restrict__ bp,
                                                    const float* __restrict__ htab,
                                                    const float* __restrict__ dtab,
                                                    float* __restrict__ x) {
    int idx = blockIdx.x * 256 + threadIdx.x;
    if (idx >= Mc * Dc) return;
    int m = idx / Dc, d = idx - m * Dc;
    x[idx] = pm25[m] * Wp[d] + bp[d] + htab[(size_t)hour[m] * Dc + d] +
             dtab[(size_t)day[m] * Dc + d];
}

// ---------------- layernorm (fp32 in -> bf16 out) ----------------
__global__ __launch_bounds__(128) void ln_kernel(const float* __restrict__ x,
                                                 const float* __restrict__ g,
                                                 const float* __restrict__ bb,
                                                 __bf16* __restrict__ out) {
    int t = blockIdx.x;
    const float* xr = x + (size_t)t * Dc;
    int tid = threadIdx.x;
    float v0 = xr[tid], v1 = xr[tid + 128], v2 = xr[tid + 256];
    float s = v0 + v1 + v2, s2 = v0 * v0 + v1 * v1 + v2 * v2;
    s = wave32_sum(s); s2 = wave32_sum(s2);
    __shared__ float sh1[4], sh2[4];
    int wid = tid >> 5, lane = tid & 31;
    if (lane == 0) { sh1[wid] = s; sh2[wid] = s2; }
    __syncthreads();
    s  = sh1[0] + sh1[1] + sh1[2] + sh1[3];
    s2 = sh2[0] + sh2[1] + sh2[2] + sh2[3];
    float mean = s * (1.f / Dc);
    float var  = s2 * (1.f / Dc) - mean * mean;
    float inv  = rsqrtf(var + 1e-5f);
    __bf16* orow = out + (size_t)t * Dc;
    orow[tid]       = (__bf16)((v0 - mean) * inv * g[tid]       + bb[tid]);
    orow[tid + 128] = (__bf16)((v1 - mean) * inv * g[tid + 128] + bb[tid + 128]);
    orow[tid + 256] = (__bf16)((v2 - mean) * inv * g[tid + 256] + bb[tid + 256]);
}

// ---------------- final LN + head ----------------
__global__ __launch_bounds__(128) void head_kernel(const float* __restrict__ x,
                                                   const float* __restrict__ g,
                                                   const float* __restrict__ bb,
                                                   const float* __restrict__ wh,
                                                   const float* __restrict__ bh,
                                                   float* __restrict__ out) {
    int t = blockIdx.x;
    const float* xr = x + (size_t)t * Dc;
    int tid = threadIdx.x;
    float v0 = xr[tid], v1 = xr[tid + 128], v2 = xr[tid + 256];
    float s = v0 + v1 + v2, s2 = v0 * v0 + v1 * v1 + v2 * v2;
    s = wave32_sum(s); s2 = wave32_sum(s2);
    __shared__ float sh1[4], sh2[4], sh3[4];
    int wid = tid >> 5, lane = tid & 31;
    if (lane == 0) { sh1[wid] = s; sh2[wid] = s2; }
    __syncthreads();
    s  = sh1[0] + sh1[1] + sh1[2] + sh1[3];
    s2 = sh2[0] + sh2[1] + sh2[2] + sh2[3];
    float mean = s * (1.f / Dc);
    float var  = s2 * (1.f / Dc) - mean * mean;
    float inv  = rsqrtf(var + 1e-5f);
    float a = ((v0 - mean) * inv * g[tid]       + bb[tid])       * wh[tid] +
              ((v1 - mean) * inv * g[tid + 128] + bb[tid + 128]) * wh[tid + 128] +
              ((v2 - mean) * inv * g[tid + 256] + bb[tid + 256]) * wh[tid + 256];
    a = wave32_sum(a);
    if (lane == 0) sh3[wid] = a;
    __syncthreads();
    if (tid == 0) out[t] = sh3[0] + sh3[1] + sh3[2] + sh3[3] + bh[0];
}

// ---------------- WMMA GEMM: C[M,N] = A[M,K](bf16) x B[K,N], B stored transposed BT[N][K] ----------------
// Register-blocked: each wave computes a 32x64 output tile (2 M-subtiles x 4 N-subtiles,
// 8 f32x8 accumulators). Per K=32 step: 2 A-frags + 4 B-frags feed 8 WMMAs.
// MODE 0: store bf16 row-major          MODE 1: +bias, ReLU, store bf16
// MODE 2: fp32 residual: C += acc+bias  MODE 3: V transpose store -> VT[B,H,64,T]
template <int MODE>
__global__ __launch_bounds__(128) void gemm_bf16(const __bf16* __restrict__ A, int lda,
                                                 const __bf16* __restrict__ BT, int ldb,
                                                 int M, int N, int K,
                                                 __bf16* __restrict__ Obf,
                                                 float* __restrict__ Of32,
                                                 int ldc,
                                                 const float* __restrict__ bias) {
    int wid = threadIdx.x >> 5, lane = threadIdx.x & 31;
    int tile = blockIdx.x * 4 + wid;
    int tilesN = N >> 6;                      // 64-wide N tiles
    int tM = tile / tilesN, tN = tile - tM * tilesN;
    if (tM >= (M >> 5)) return;               // wave-uniform, EXEC stays full
    int half = lane >> 4, l16 = lane & 15;

    const __bf16* arow0 = A + (size_t)(tM * 32 + l16) * lda;
    const __bf16* arow1 = arow0 + (size_t)16 * lda;
    const __bf16* brow  = BT + (size_t)(tN * 64 + l16) * ldb;

    f32x8 acc[2][4];
#pragma unroll
    for (int mt = 0; mt < 2; ++mt)
#pragma unroll
        for (int j = 0; j < 4; ++j) acc[mt][j] = zero8();

    for (int k = 0; k < K; k += 32) {
        bf16x16 af0 = load_afrag(arow0 + k, half);
        bf16x16 af1 = load_afrag(arow1 + k, half);
#pragma unroll
        for (int j = 0; j < 4; ++j) {
            bf16x16 bfr = load_bfrag(brow + (size_t)(j * 16) * ldb + k, half);
            acc[0][j] = wmma_bf16(af0, bfr, acc[0][j]);
            acc[1][j] = wmma_bf16(af1, bfr, acc[1][j]);
        }
    }

#pragma unroll
    for (int mt = 0; mt < 2; ++mt) {
        int rbase = tM * 32 + mt * 16 + half * 8;
#pragma unroll
        for (int j = 0; j < 4; ++j) {
            int col = tN * 64 + j * 16 + l16;
#pragma unroll
            for (int r = 0; r < 8; ++r) {
                int row = rbase + r;
                float v = acc[mt][j][r];
                if (MODE == 0) {
                    Obf[(size_t)row * ldc + col] = (__bf16)v;
                } else if (MODE == 1) {
                    v += bias[col];
                    Obf[(size_t)row * ldc + col] = (__bf16)(v > 0.f ? v : 0.f);
                } else if (MODE == 2) {
                    Of32[(size_t)row * ldc + col] += v + bias[col];
                } else {  // MODE 3: row = b*T + t ; col = h*64 + c ; VT[((b*H+h)*64+c)*T + t]
                    int bb = row >> 9, tt = row & 511;
                    Obf[(((size_t)bb * Hc + (col >> 6)) * 64 + (col & 63)) * Tc + tt] = (__bf16)v;
                }
            }
        }
    }
}

// ---------------- flash attention: 1 wave per 16 query rows ----------------
__global__ __launch_bounds__(128) void attn_kernel(const __bf16* __restrict__ Q,
                                                   const __bf16* __restrict__ Kb,
                                                   const __bf16* __restrict__ VT,
                                                   __bf16* __restrict__ att) {
    __shared__ __bf16 ldsP[4][16 * 32];   // per-wave P tile (D-layout -> A-layout transpose)
    int wid = threadIdx.x >> 5, lane = threadIdx.x & 31;
    int gw = blockIdx.x * 4 + wid;
    if (gw >= Bc * Hc * (Tc / 16)) return;
    int qb = gw & 31;
    int h  = (gw >> 5) & 7;
    int b  = gw >> 8;
    int half = lane >> 4, l16 = lane & 15;
    __bf16* P = ldsP[wid];

    const __bf16* qrow = Q + ((size_t)(b * Tc + qb * 16 + l16) * NQK + h * HSP);
    bf16x16 qf0 = load_afrag(qrow, half);
    bf16x16 qf1 = load_afrag(qrow + 32, half);

    f32x8 o0 = zero8(), o1 = zero8(), o2 = zero8();
    float mi[8], li[8];
#pragma unroll
    for (int r = 0; r < 8; ++r) { mi[r] = -1e30f; li[r] = 0.f; }

    const float scale = 0.14433756729740643f;  // 1/sqrt(48)
    int rbase = qb * 16 + half * 8;
    int jmax = (qb * 16 + 15) >> 5;

    for (int j = 0; j <= jmax; ++j) {
        int s0 = j << 5;
        const __bf16* ka  = Kb + ((size_t)(b * Tc + s0 + l16) * NQK + h * HSP);
        const __bf16* kb2 = ka + (size_t)16 * NQK;
        f32x8 sa = zero8(), sb = zero8();
        sa = wmma_bf16(qf0, load_bfrag(ka, half), sa);
        sa = wmma_bf16(qf1, load_bfrag(ka + 32, half), sa);
        sb = wmma_bf16(qf0, load_bfrag(kb2, half), sb);
        sb = wmma_bf16(qf1, load_bfrag(kb2 + 32, half), sb);

        float pa[8], pb[8];
#pragma unroll
        for (int r = 0; r < 8; ++r) {
            int rowg = rbase + r;
            float xa = (s0 + l16      <= rowg) ? sa[r] * scale : -1e30f;
            float xb = (s0 + 16 + l16 <= rowg) ? sb[r] * scale : -1e30f;
            float tmax  = half16_max(fmaxf(xa, xb));
            float mnew  = fmaxf(mi[r], tmax);
            float alpha = __expf(mi[r] - mnew);
            float ea = __expf(xa - mnew), eb = __expf(xb - mnew);
            float psum = half16_sum(ea + eb);
            li[r] = li[r] * alpha + psum;
            mi[r] = mnew;
            o0[r] *= alpha; o1[r] *= alpha; o2[r] *= alpha;
            pa[r] = ea; pb[r] = eb;
        }
        int prow = half * 8;
#pragma unroll
        for (int r = 0; r < 8; ++r) {
            P[(prow + r) * 32 + l16]      = (__bf16)pa[r];
            P[(prow + r) * 32 + 16 + l16] = (__bf16)pb[r];
        }
        bf16x16 pf = load_afrag(P + l16 * 32, half);
        const __bf16* vbase = VT + (((size_t)(b * Hc + h) * 64 + l16) * Tc) + s0;
        o0 = wmma_bf16(pf, load_bfrag(vbase, half), o0);
        o1 = wmma_bf16(pf, load_bfrag(vbase + (size_t)16 * Tc, half), o1);
        o2 = wmma_bf16(pf, load_bfrag(vbase + (size_t)32 * Tc, half), o2);
    }
#pragma unroll
    for (int r = 0; r < 8; ++r) {
        float inv = 1.f / li[r];
        int rowg = rbase + r;
        __bf16* arow = att + (size_t)(b * Tc + rowg) * Dc + h * HSc;
        arow[l16]      = (__bf16)(o0[r] * inv);
        arow[16 + l16] = (__bf16)(o1[r] * inv);
        arow[32 + l16] = (__bf16)(o2[r] * inv);
    }
}

// ---------------- host launcher ----------------
extern "C" void kernel_launch(void* const* d_in, const int* in_sizes, int n_in,
                              void* d_out, int out_size, void* d_ws, size_t ws_size,
                              hipStream_t stream) {
    const float* pm25     = (const float*)d_in[0];
    const int*   hour_idx = (const int*)d_in[1];
    const int*   day_idx  = (const int*)d_in[2];
    const float* W_pm25   = (const float*)d_in[3];
    const float* b_pm25   = (const float*)d_in[4];
    const float* hour_tab = (const float*)d_in[5];
    const float* day_tab  = (const float*)d_in[6];
    const float* ln1_g = (const float*)d_in[7];
    const float* ln1_b = (const float*)d_in[8];
    const float* ln3_g = (const float*)d_in[9];
    const float* ln3_b = (const float*)d_in[10];
    const float* Wq = (const float*)d_in[11];
    const float* Wk = (const float*)d_in[12];
    const float* Wv = (const float*)d_in[13];
    const float* Wo = (const float*)d_in[14];
    const float* bo = (const float*)d_in[15];
    const float* W1 = (const float*)d_in[16];
    const float* b1 = (const float*)d_in[17];
    const float* W2 = (const float*)d_in[18];
    const float* b2 = (const float*)d_in[19];
    const float* lnf_g  = (const float*)d_in[20];
    const float* lnf_b  = (const float*)d_in[21];
    const float* W_head = (const float*)d_in[22];
    const float* b_head = (const float*)d_in[23];
    float* out = (float*)d_out;

    // ---- workspace carve (~174 MB total) ----
    char* base = (char*)d_ws;
    size_t off = 0;
    auto alloc = [&](size_t bytes) -> void* {
        void* p = base + off;
        off += (bytes + 255) & ~(size_t)255;
        return p;
    };
    float*  x     = (float*)alloc((size_t)Mc * Dc * 4);
    __bf16* hbuf  = (__bf16*)alloc((size_t)Mc * Dc * 2);
    __bf16* qbuf  = (__bf16*)alloc((size_t)Mc * NQK * 2);
    __bf16* kbuf  = (__bf16*)alloc((size_t)Mc * NQK * 2);
    __bf16* vtbuf = (__bf16*)alloc((size_t)Bc * Hc * HSP * Tc * 2);
    __bf16* attb  = (__bf16*)alloc((size_t)Mc * Dc * 2);
    __bf16* ffb   = (__bf16*)alloc((size_t)Mc * FFc * 2);
    __bf16* wqT   = (__bf16*)alloc((size_t)Lc * NQK * Dc * 2);
    __bf16* wkT   = (__bf16*)alloc((size_t)Lc * NQK * Dc * 2);
    __bf16* wvT   = (__bf16*)alloc((size_t)Lc * NQK * Dc * 2);
    __bf16* woT   = (__bf16*)alloc((size_t)Lc * Dc * Dc * 2);
    __bf16* w1T   = (__bf16*)alloc((size_t)Lc * FFc * Dc * 2);
    __bf16* w2T   = (__bf16*)alloc((size_t)Lc * Dc * FFc * 2);
    (void)ws_size; (void)in_sizes; (void)n_in; (void)out_size;

    // ---- weight conversion (fp32 -> transposed bf16) ----
    for (int l = 0; l < Lc; ++l) {
        int nQ = NQK * Dc;
        convQKV_kernel<<<(nQ + 255) / 256, 256, 0, stream>>>(Wq + (size_t)l * Hc * Dc * HSc, wqT + (size_t)l * nQ);
        convQKV_kernel<<<(nQ + 255) / 256, 256, 0, stream>>>(Wk + (size_t)l * Hc * Dc * HSc, wkT + (size_t)l * nQ);
        convQKV_kernel<<<(nQ + 255) / 256, 256, 0, stream>>>(Wv + (size_t)l * Hc * Dc * HSc, wvT + (size_t)l * nQ);
        convT_kernel<<<(Dc * Dc + 255) / 256, 256, 0, stream>>>(Wo + (size_t)l * Dc * Dc, woT + (size_t)l * Dc * Dc, Dc, Dc);
        convT_kernel<<<(Dc * FFc + 255) / 256, 256, 0, stream>>>(W1 + (size_t)l * Dc * FFc, w1T + (size_t)l * FFc * Dc, Dc, FFc);
        convT_kernel<<<(FFc * Dc + 255) / 256, 256, 0, stream>>>(W2 + (size_t)l * FFc * Dc, w2T + (size_t)l * Dc * FFc, FFc, Dc);
    }

    // ---- embedding ----
    embed_kernel<<<(Mc * Dc + 255) / 256, 256, 0, stream>>>(pm25, hour_idx, day_idx, W_pm25,
                                                            b_pm25, hour_tab, day_tab, x);

    const int tilesM32 = Mc / 32;  // 512
    // ---- transformer layers ----
    for (int l = 0; l < Lc; ++l) {
        ln_kernel<<<Mc, 128, 0, stream>>>(x, ln1_g + l * Dc, ln1_b + l * Dc, hbuf);

        int blkQKV = (tilesM32 * (NQK / 64) + 3) / 4;
        gemm_bf16<0><<<blkQKV, 128, 0, stream>>>(hbuf, Dc, wqT + (size_t)l * NQK * Dc, Dc,
                                                 Mc, NQK, Dc, qbuf, nullptr, NQK, nullptr);
        gemm_bf16<0><<<blkQKV, 128, 0, stream>>>(hbuf, Dc, wkT + (size_t)l * NQK * Dc, Dc,
                                                 Mc, NQK, Dc, kbuf, nullptr, NQK, nullptr);
        gemm_bf16<3><<<blkQKV, 128, 0, stream>>>(hbuf, Dc, wvT + (size_t)l * NQK * Dc, Dc,
                                                 Mc, NQK, Dc, vtbuf, nullptr, 0, nullptr);

        attn_kernel<<<Bc * Hc * (Tc / 16) / 4, 128, 0, stream>>>(qbuf, kbuf, vtbuf, attb);

        int blkWo = (tilesM32 * (Dc / 64) + 3) / 4;
        gemm_bf16<2><<<blkWo, 128, 0, stream>>>(attb, Dc, woT + (size_t)l * Dc * Dc, Dc,
                                                Mc, Dc, Dc, nullptr, x, Dc, bo + l * Dc);

        ln_kernel<<<Mc, 128, 0, stream>>>(x, ln3_g + l * Dc, ln3_b + l * Dc, hbuf);

        int blkF1 = (tilesM32 * (FFc / 64) + 3) / 4;
        gemm_bf16<1><<<blkF1, 128, 0, stream>>>(hbuf, Dc, w1T + (size_t)l * FFc * Dc, Dc,
                                                Mc, FFc, Dc, ffb, nullptr, FFc, b1 + l * FFc);
        int blkF2 = (tilesM32 * (Dc / 64) + 3) / 4;
        gemm_bf16<2><<<blkF2, 128, 0, stream>>>(ffb, FFc, w2T + (size_t)l * Dc * FFc, FFc,
                                                Mc, Dc, FFc, nullptr, x, Dc, b2 + l * Dc);
    }

    // ---- final LN + head ----
    head_kernel<<<Mc, 128, 0, stream>>>(x, lnf_g, lnf_b, W_head, b_head, out);
}
